// AtomTransformer_52862457480070
// MI455X (gfx1250) — compile-verified, matrix operand order checked by reading, not run on hardware
//
#include <hip/hip_runtime.h>

typedef _Float16 half_t;
typedef __attribute__((ext_vector_type(16))) _Float16 v16h;
typedef __attribute__((ext_vector_type(8)))  _Float16 v8h;
typedef __attribute__((ext_vector_type(8)))  float    v8f;

// ---------------------------------------------------------------------------
// WMMA helpers (CDNA5 wave32, v_wmma_f32_16x16x32_f16)
// A-matrix: lane a(0-15)=row a holds K {0..7,16..23}; lane a+16=row a, K {8..15,24..31}
// B-matrix: lane n(0-15)=col n, same K split.  C/D f32: elem v -> row v+8*(lane>=16), col lane&15
// ---------------------------------------------------------------------------
static __device__ __forceinline__ v8f wmma32(v16h a, v16h b, v8f c) {
  return __builtin_amdgcn_wmma_f32_16x16x32_f16(false, a, false, b, (short)0, c,
                                                false, false);
}

// Load an A/B fragment from memory where this lane's row/col is contiguous in K.
static __device__ __forceinline__ v16h frag_ld(const half_t* rowK, int hi) {
  v8h a0 = *(const v8h*)(rowK + hi * 8);
  v8h a1 = *(const v8h*)(rowK + hi * 8 + 16);
  v16h r;
#pragma unroll
  for (int e = 0; e < 8; ++e) { r[e] = a0[e]; r[e + 8] = a1[e]; }
  return r;
}

// CDNA5 async global->LDS copy (16B per lane), tracked by ASYNCcnt.
static __device__ __forceinline__ void async_b128(void* lds_ptr, const void* gptr) {
  unsigned lds_off = (unsigned)(size_t)lds_ptr;  // low 32 bits = LDS byte offset
  asm volatile("global_load_async_to_lds_b128 %0, %1, off"
               :: "v"(lds_off), "v"(gptr) : "memory");
}
static __device__ __forceinline__ void wait_async0() {
  asm volatile("s_wait_asynccnt 0x0" ::: "memory");
}

// ---------------------------------------------------------------------------
// fp32 -> fp16 weight conversion
// ---------------------------------------------------------------------------
__global__ void k_cvt(const float* __restrict__ s, half_t* __restrict__ d, int n) {
  int i = blockIdx.x * 256 + threadIdx.x;
  if (i < n) d[i] = (half_t)s[i];
}

// ---------------------------------------------------------------------------
// Embedding: x = concat(feat[6], atom_emb[idx+1][410], hyb@Wh^T[64],
//                       donac@Wd^T[16], spin@Ws^T[16])  -> fp32 + fp16
// ---------------------------------------------------------------------------
__global__ void k_embed(const int* __restrict__ atom_idx, const float* __restrict__ hyb,
                        const float* __restrict__ donac, const float* __restrict__ spin,
                        const float* __restrict__ feat, const float* __restrict__ atom_emb,
                        const float* __restrict__ w_hyb, const float* __restrict__ w_donac,
                        const float* __restrict__ w_spin,
                        float* __restrict__ x, half_t* __restrict__ xh) {
  int row = blockIdx.x;                  // b*256+n
  int aid = atom_idx[row] + 1;
  for (int d = threadIdx.x; d < 512; d += blockDim.x) {
    float v;
    if (d < 6) {
      v = feat[row * 6 + d];
    } else if (d < 416) {
      v = atom_emb[aid * 410 + (d - 6)];
    } else if (d < 480) {
      int o = d - 416; float s = 0.f;
#pragma unroll
      for (int c = 0; c < 7; ++c) s += hyb[row * 7 + c] * w_hyb[o * 7 + c];
      v = s;
    } else if (d < 496) {
      int o = d - 480;
      v = donac[row * 2] * w_donac[o * 2] + donac[row * 2 + 1] * w_donac[o * 2 + 1];
    } else {
      int o = d - 496;
      v = spin[row * 2] * w_spin[o * 2] + spin[row * 2 + 1] * w_spin[o * 2 + 1];
    }
    x[(size_t)row * 512 + d]  = v;
    xh[(size_t)row * 512 + d] = (half_t)v;
  }
}

// ---------------------------------------------------------------------------
// Generic WMMA GEMM: C[M,N] = act( A[M,K](f16) @ W[N,K]^T(f16) + bias + resid )
// block = 256 thr (8 waves); block tile 256(M) x 64(N); wave tile 32x64.
// W tiles double-buffered in LDS via global_load_async_to_lds_b128 (ASYNCcnt).
// ---------------------------------------------------------------------------
__global__ void __launch_bounds__(256)
k_gemm(const half_t* __restrict__ A, const half_t* __restrict__ W,
       const float* __restrict__ bias, const float* __restrict__ resid,
       half_t* __restrict__ Ch, float* __restrict__ Cf,
       int M, int N, int K, int relu) {
  __shared__ half_t sW[2][64 * 32];
  const int wave = threadIdx.x >> 5;
  const int lane = threadIdx.x & 31;
  const int r = lane & 15, hi = lane >> 4;
  const int m0 = blockIdx.x * 256 + wave * 32;
  const int n0 = blockIdx.y * 64;

  v8f acc[2][4];
#pragma unroll
  for (int g2 = 0; g2 < 2; ++g2)
#pragma unroll
    for (int j = 0; j < 4; ++j)
#pragma unroll
      for (int v = 0; v < 8; ++v) acc[g2][j][v] = 0.f;

  const int nrow = threadIdx.x >> 2;        // 0..63
  const int kcol = (threadIdx.x & 3) * 8;   // 0,8,16,24
  const half_t* wsrc  = W + (size_t)(n0 + nrow) * K + kcol;
  const half_t* arow0 = A + (size_t)(m0 + r) * K;
  const half_t* arow1 = A + (size_t)(m0 + 16 + r) * K;

  // prologue: stage tile 0
  async_b128(&sW[0][nrow * 32 + kcol], wsrc);
  wait_async0();
  __syncthreads();

  int buf = 0;
  for (int k0 = 0; k0 < K; k0 += 32) {
    const int nxt = k0 + 32;
    if (nxt < K) {  // overlap next W tile copy with this tile's WMMAs
      async_b128(&sW[buf ^ 1][nrow * 32 + kcol], wsrc + nxt);
      __builtin_prefetch(arow0 + nxt, 0, 1);  // global_prefetch_b8
      __builtin_prefetch(arow1 + nxt, 0, 1);
    }
    v16h a0 = frag_ld(arow0 + k0, hi);
    v16h a1 = frag_ld(arow1 + k0, hi);
#pragma unroll
    for (int j = 0; j < 4; ++j) {
      v16h bf = frag_ld(&sW[buf][(j * 16 + r) * 32], hi);
      acc[0][j] = wmma32(a0, bf, acc[0][j]);
      acc[1][j] = wmma32(a1, bf, acc[1][j]);
    }
    if (nxt < K) wait_async0();
    __syncthreads();
    buf ^= 1;
  }

#pragma unroll
  for (int g2 = 0; g2 < 2; ++g2) {
#pragma unroll
    for (int j = 0; j < 4; ++j) {
      int col = n0 + j * 16 + r;
      float bcol = bias ? bias[col] : 0.f;
#pragma unroll
      for (int v = 0; v < 8; ++v) {
        int row = m0 + g2 * 16 + hi * 8 + v;
        float val = acc[g2][j][v] + bcol;
        if (resid) val += resid[(size_t)row * N + col];
        if (relu)  val = fmaxf(val, 0.f);
        if (Ch) Ch[(size_t)row * N + col] = (half_t)val;
        if (Cf) Cf[(size_t)row * N + col] = val;
      }
    }
  }
}

// ---------------------------------------------------------------------------
// Attention (flash-style, transposed scores): per (b,h) block of 8 waves.
// S^T = K @ Q^T via WMMA; bias/softmax in f32; P converts in-lane to the
// B-fragment for  O^T = V^T @ P^T  WMMA.  K/V staged in LDS via async copies.
// ---------------------------------------------------------------------------
__global__ void __launch_bounds__(256)
k_attn(const half_t* __restrict__ qkv, const float* __restrict__ pdist,
       const float* __restrict__ angle, const float* __restrict__ adjm,
       const int* __restrict__ mask,
       const float* __restrict__ gamma_p, const float* __restrict__ gamma_adj,
       const float* __restrict__ w_bias, half_t* __restrict__ att) {
  __shared__ half_t sK[256 * 32];
  __shared__ half_t sV[256 * 32];
  const int bh = blockIdx.x;
  const int b = bh >> 4, h = bh & 15;
  const size_t qbase = (size_t)b * 256 * 1536;

  {  // async cooperative K/V stage: thread t loads seq row t (32 halves each)
    int t = threadIdx.x;
    const half_t* ksrc = qkv + qbase + (size_t)t * 1536 + 512 + h * 32;
    const half_t* vsrc = ksrc + 512;
#pragma unroll
    for (int c = 0; c < 4; ++c) {
      async_b128(&sK[t * 32 + c * 8], ksrc + c * 8);
      async_b128(&sV[t * 32 + c * 8], vsrc + c * 8);
    }
    wait_async0();
  }
  __syncthreads();

  const float gp = gamma_p[h], ga = gamma_adj[h];
  const float wb0 = w_bias[h * 2 + 0], wb1 = w_bias[h * 2 + 1];
  const int wave = threadIdx.x >> 5, lane = threadIdx.x & 31;
  const int r = lane & 15, hi = lane >> 4;

  for (int it = wave; it < 16; it += 8) {
    const int i0 = it * 16;
    const int i = i0 + r;                    // query index handled by this lane (col)
    v16h qf = frag_ld(qkv + qbase + (size_t)i * 1536 + h * 32, hi);  // Q^T B-frag
    v8f o0, o1;
#pragma unroll
    for (int v = 0; v < 8; ++v) { o0[v] = 0.f; o1[v] = 0.f; }
    float m = -1e30f, l = 0.f;
    const size_t pbase = (size_t)(b * 256 + i) * 256;

    for (int j0 = 0; j0 < 256; j0 += 32) {   // two 16-wide key tiles per step
      float p[2][8];
      float tm = -1e30f;
#pragma unroll
      for (int s2 = 0; s2 < 2; ++s2) {
        const int js = j0 + s2 * 16;
        v16h kf = frag_ld(&sK[(js + r) * 32], hi);   // K rows as A-frag
        v8f z;
#pragma unroll
        for (int v = 0; v < 8; ++v) z[v] = 0.f;
        v8f st = wmma32(kf, qf, z);                  // S^T tile: row j, col i
        const int jb = js + hi * 8;
#pragma unroll
        for (int v = 0; v < 8; ++v) {
          const int j = jb + v;
          float sc = st[v] * 0.17677669529663687f;   // 1/sqrt(32)
          sc -= gp * pdist[pbase + j];
          sc += wb0 * angle[(pbase + j) * 2] + wb1 * angle[(pbase + j) * 2 + 1];
          sc += ga * adjm[pbase + j];
          if (mask[pbase + j] != 0) sc = -1e9f;
          p[s2][v] = sc;
          tm = fmaxf(tm, sc);
        }
      }
      tm = fmaxf(tm, __shfl_xor(tm, 16));            // combine lane halves (per col i)
      const float mn = fmaxf(m, tm);
      const float corr = __expf(m - mn);
      m = mn;
      l *= corr;
#pragma unroll
      for (int v = 0; v < 8; ++v) { o0[v] *= corr; o1[v] *= corr; }

      v16h pb;                                       // P^T as B-frag (in-lane repack)
#pragma unroll
      for (int v = 0; v < 8; ++v) {
        const float e0 = __expf(p[0][v] - m);
        const float e1 = __expf(p[1][v] - m);
        l += e0 + e1;
        pb[v]     = (half_t)e0;                      // K = hi*8 + v
        pb[8 + v] = (half_t)e1;                      // K = 16 + hi*8 + v
      }
      v16h vf0, vf1;                                 // V^T A-frags (rows d, K=j)
#pragma unroll
      for (int e = 0; e < 16; ++e) {
        const int jrel = ((e < 8) ? e : e + 8) + hi * 8;
        const int j = j0 + jrel;
        vf0[e] = sV[j * 32 + r];
        vf1[e] = sV[j * 32 + 16 + r];
      }
      o0 = wmma32(vf0, pb, o0);
      o1 = wmma32(vf1, pb, o1);
    }
    l += __shfl_xor(l, 16);
    const float inv = 1.f / l;
    half_t* dst = att + (size_t)(b * 256 + i) * 512 + h * 32;
    v8h r0, r1;
#pragma unroll
    for (int v = 0; v < 8; ++v) {
      r0[v] = (half_t)(o0[v] * inv);
      r1[v] = (half_t)(o1[v] * inv);
    }
    *(v8h*)(dst + hi * 8)      = r0;   // d = hi*8 + v
    *(v8h*)(dst + 16 + hi * 8) = r1;   // d = 16 + hi*8 + v
  }
}

// ---------------------------------------------------------------------------
// Row LayerNorm over D=512: x,xh = LN(y)
// ---------------------------------------------------------------------------
__global__ void __launch_bounds__(256)
k_ln(const float* __restrict__ y, const float* __restrict__ g,
     const float* __restrict__ bb, float* __restrict__ x, half_t* __restrict__ xh) {
  __shared__ float red[256];
  const int row = blockIdx.x, t = threadIdx.x;
  const float* yr = y + (size_t)row * 512;
  float v0 = yr[t], v1 = yr[t + 256];
  red[t] = v0 + v1; __syncthreads();
  for (int s = 128; s > 0; s >>= 1) { if (t < s) red[t] += red[t + s]; __syncthreads(); }
  const float mean = red[0] * (1.f / 512.f);
  __syncthreads();
  const float d0 = v0 - mean, d1 = v1 - mean;
  red[t] = d0 * d0 + d1 * d1; __syncthreads();
  for (int s = 128; s > 0; s >>= 1) { if (t < s) red[t] += red[t + s]; __syncthreads(); }
  const float rstd = rsqrtf(red[0] * (1.f / 512.f) + 1e-5f);
  const float o0 = g[t] * (d0 * rstd) + bb[t];
  const float o1 = g[t + 256] * (d1 * rstd) + bb[t + 256];
  float* xr = x + (size_t)row * 512;
  half_t* xhr = xh + (size_t)row * 512;
  xr[t] = o0; xr[t + 256] = o1;
  xhr[t] = (half_t)o0; xhr[t + 256] = (half_t)o1;
}

// ---------------------------------------------------------------------------
// Pooling: f[b] = [mean, max, min] over N with out_mask
// ---------------------------------------------------------------------------
__global__ void __launch_bounds__(256)
k_pool(const float* __restrict__ x, const int* __restrict__ om,
       const float* __restrict__ n_atoms, float* __restrict__ f) {
  const int b = blockIdx.x;
  const float inv_n = 1.f / n_atoms[b];
  for (int d = threadIdx.x; d < 512; d += 256) {
    float s = 0.f, mx = -3.4e38f, mn = 3.4e38f;
    for (int n = 0; n < 256; ++n) {
      const float v = x[(size_t)(b * 256 + n) * 512 + d];
      const int msk = om[b * 256 + n];
      const float big = msk ? 0.f : 1e4f;
      s += msk ? v : 0.f;
      mx = fmaxf(mx, v - big);
      mn = fminf(mn, v + big);
    }
    f[(size_t)b * 1536 + d]        = s * inv_n;
    f[(size_t)b * 1536 + 512 + d]  = mx;
    f[(size_t)b * 1536 + 1024 + d] = mn;
  }
}

// ---------------------------------------------------------------------------
// 3 scalar heads: out = relu(f @ w1^T + b1) @ w2^T + b2   (gap, s1, t1)
// ---------------------------------------------------------------------------
__global__ void k_heads(const float* __restrict__ f,
                        const float* w1a, const float* b1a, const float* w2a, const float* b2a,
                        const float* w1b, const float* b1b, const float* w2b, const float* b2b,
                        const float* w1c, const float* b1c, const float* w2c, const float* b2c,
                        float* __restrict__ out) {
  __shared__ float part[48];
  const int b = blockIdx.x;
  const int t = threadIdx.x, head = t >> 4, j = t & 15;
  const float* w1 = head == 0 ? w1a : head == 1 ? w1b : w1c;
  const float* b1 = head == 0 ? b1a : head == 1 ? b1b : b1c;
  const float* w2 = head == 0 ? w2a : head == 1 ? w2b : w2c;
  const float* fb = f + (size_t)b * 1536;
  float s = 0.f;
  for (int d = 0; d < 1536; ++d) s += fb[d] * w1[j * 1536 + d];
  s = fmaxf(s + b1[j], 0.f) * w2[j];
  part[t] = s; __syncthreads();
  if (j == 0) {
    float acc = 0.f;
    for (int jj = 0; jj < 16; ++jj) acc += part[head * 16 + jj];
    const float* b2 = head == 0 ? b2a : head == 1 ? b2b : b2c;
    out[head * 32 + b] = acc + b2[0];
  }
}

// ---------------------------------------------------------------------------
// Host orchestration
// ---------------------------------------------------------------------------
extern "C" void kernel_launch(void* const* d_in, const int* in_sizes, int n_in,
                              void* d_out, int out_size, void* d_ws, size_t ws_size,
                              hipStream_t stream) {
  (void)n_in; (void)out_size; (void)ws_size;
  const int L = 6, H = 16, D = 512, DFF = 2048, B = 32, N = 256, M = B * N;

  const float *hyb, *donac, *spin, *feat, *pdist, *angle, *adj, *n_atoms;
  const int *atom_idx, *mask, *out_mask;
  const float *atom_emb, *w_hyb, *w_donac, *w_spin;
  const float *gamma_p[6], *gamma_adj[6], *w_bias[6], *w_att[6], *w_ff[6], *b_ff[6],
              *ln1_g[6], *ln1_b[6], *w_l1[6], *b_l1[6], *w_l2[6], *b_l2[6],
              *ln2_g[6], *ln2_b[6];
  const float *hw1[3], *hb1[3], *hw2[3], *hb2[3];

  auto gf = [&](int i) { return (const float*)d_in[i]; };
  auto gi = [&](int i) { return (const int*)d_in[i]; };

  const bool insertion_order = (in_sizes[0] == B * N);  // atom_idx first vs adj first
  if (insertion_order) {
    atom_idx = gi(0); hyb = gf(1); donac = gf(2); spin = gf(3); feat = gf(4);
    pdist = gf(5); angle = gf(6); adj = gf(7); mask = gi(8); out_mask = gi(9);
    n_atoms = gf(10);
    int p = 11;
    for (int l = 0; l < L; ++l) {
      gamma_p[l] = gf(p + 0); gamma_adj[l] = gf(p + 1); w_bias[l] = gf(p + 2);
      w_att[l] = gf(p + 3);   w_ff[l] = gf(p + 4);      b_ff[l] = gf(p + 5);
      ln1_g[l] = gf(p + 6);   ln1_b[l] = gf(p + 7);     w_l1[l] = gf(p + 8);
      b_l1[l] = gf(p + 9);    w_l2[l] = gf(p + 10);     b_l2[l] = gf(p + 11);
      ln2_g[l] = gf(p + 12);  ln2_b[l] = gf(p + 13);    p += 14;
    }
    atom_emb = gf(p++); w_hyb = gf(p++); w_donac = gf(p++); w_spin = gf(p++);
    for (int k = 0; k < 3; ++k) {
      hw1[k] = gf(p + 0); hb1[k] = gf(p + 1); hw2[k] = gf(p + 2); hb2[k] = gf(p + 3);
      p += 4;
    }
  } else {  // alphabetical pytree flattening
    adj = gf(0); angle = gf(1); atom_idx = gi(2); donac = gf(3); feat = gf(4);
    hyb = gf(5); mask = gi(6); n_atoms = gf(7); out_mask = gi(8);
    int p = 9;
    atom_emb = gf(p++);
    hb1[0] = gf(p++); hb2[0] = gf(p++); hw1[0] = gf(p++); hw2[0] = gf(p++);  // gap
    for (int l = 0; l < L; ++l) {
      b_ff[l] = gf(p++);  b_l1[l] = gf(p++);  b_l2[l] = gf(p++);
      gamma_adj[l] = gf(p++); gamma_p[l] = gf(p++);
      ln1_b[l] = gf(p++); ln1_g[l] = gf(p++); ln2_b[l] = gf(p++); ln2_g[l] = gf(p++);
      w_att[l] = gf(p++); w_bias[l] = gf(p++); w_ff[l] = gf(p++);
      w_l1[l] = gf(p++);  w_l2[l] = gf(p++);
    }
    hb1[1] = gf(p++); hb2[1] = gf(p++); hw1[1] = gf(p++); hw2[1] = gf(p++);  // s1
    hb1[2] = gf(p++); hb2[2] = gf(p++); hw1[2] = gf(p++); hw2[2] = gf(p++);  // t1
    w_donac = gf(p++); w_hyb = gf(p++); w_spin = gf(p++);
    pdist = gf(p++); spin = gf(p++);
  }

  // workspace layout (all L2-resident)
  char* ws = (char*)d_ws;
  size_t off = 0;
  auto walloc = [&](size_t bytes) {
    void* q = ws + off;
    off += (bytes + 255) & ~size_t(255);
    return q;
  };
  float*  X   = (float*)walloc((size_t)M * D * 4);
  half_t* XH  = (half_t*)walloc((size_t)M * D * 2);
  half_t* QKV = (half_t*)walloc((size_t)M * 3 * D * 2);
  half_t* ATT = (half_t*)walloc((size_t)M * D * 2);
  float*  Y   = (float*)walloc((size_t)M * D * 4);
  half_t* H1  = (half_t*)walloc((size_t)M * DFF * 2);
  float*  FP  = (float*)walloc((size_t)B * 3 * D * 4);
  half_t* WAh = (half_t*)walloc((size_t)3 * D * D * 2);
  half_t* WFh = (half_t*)walloc((size_t)D * D * 2);
  half_t* W1h = (half_t*)walloc((size_t)DFF * D * 2);
  half_t* W2h = (half_t*)walloc((size_t)D * DFF * 2);

  k_embed<<<M, 128, 0, stream>>>(atom_idx, hyb, donac, spin, feat, atom_emb,
                                 w_hyb, w_donac, w_spin, X, XH);

  for (int l = 0; l < L; ++l) {
    k_cvt<<<(3 * D * D + 255) / 256, 256, 0, stream>>>(w_att[l], WAh, 3 * D * D);
    k_cvt<<<(D * D + 255) / 256, 256, 0, stream>>>(w_ff[l], WFh, D * D);
    k_cvt<<<(DFF * D + 255) / 256, 256, 0, stream>>>(w_l1[l], W1h, DFF * D);
    k_cvt<<<(D * DFF + 255) / 256, 256, 0, stream>>>(w_l2[l], W2h, D * DFF);

    // qkv = x @ w_att^T
    k_gemm<<<dim3(M / 256, (3 * D) / 64), 256, 0, stream>>>(
        XH, WAh, nullptr, nullptr, QKV, nullptr, M, 3 * D, D, 0);
    // attention
    k_attn<<<B * H, 256, 0, stream>>>(QKV, pdist, angle, adj, mask,
                                      gamma_p[l], gamma_adj[l], w_bias[l], ATT);
    // y = x + att @ w_ff^T + b_ff ; x = LN1(y)
    k_gemm<<<dim3(M / 256, D / 64), 256, 0, stream>>>(
        ATT, WFh, b_ff[l], X, nullptr, Y, M, D, D, 0);
    k_ln<<<M, 256, 0, stream>>>(Y, ln1_g[l], ln1_b[l], X, XH);
    // h1 = relu(x @ w_l1^T + b_l1)
    k_gemm<<<dim3(M / 256, DFF / 64), 256, 0, stream>>>(
        XH, W1h, b_l1[l], nullptr, H1, nullptr, M, DFF, D, 1);
    // y = x + h1 @ w_l2^T + b_l2 ; x = LN2(y)
    k_gemm<<<dim3(M / 256, D / 64), 256, 0, stream>>>(
        H1, W2h, b_l2[l], X, nullptr, Y, M, D, DFF, 0);
    k_ln<<<M, 256, 0, stream>>>(Y, ln2_g[l], ln2_b[l], X, XH);
  }

  k_pool<<<B, 256, 0, stream>>>(X, out_mask, n_atoms, FP);
  k_heads<<<B, 48, 0, stream>>>(FP,
                                hw1[0], hb1[0], hw2[0], hb2[0],
                                hw1[1], hb1[1], hw2[1], hb2[1],
                                hw1[2], hb1[2], hw2[2], hb2[2],
                                (float*)d_out);
}
